// TransformerDSOEncoder_83124797046900
// MI455X (gfx1250) — matrix operand order repaired and checked by reading, not compiled, not run on hardware
//
#include <hip/hip_runtime.h>
#include <hip/hip_bf16.h>

// Problem constants (match reference)
#define B_     16
#define S_     512
#define D_     512
#define H_     8
#define L_     6
#define DPH_   64
#define DFF_   2048
#define VOCAB_ 1024
#define NPC_   4096
#define NS_    4096
#define M_     (B_ * S_)   // 8192 rows in the activation matrix

typedef __attribute__((ext_vector_type(16))) _Float16 v16h;
typedef __attribute__((ext_vector_type(8)))  _Float16 v8h;
typedef __attribute__((ext_vector_type(8)))  float    v8f;

// Hout layout modes for the GEMM epilogue
#define HM_NONE 0
#define HM_ROW  1   // [m, n] row-major with leading dim ldh
#define HM_QK   2   // [b, h, s, dph]  (m=b*S+s, n=h*DPH+dph)
#define HM_VT   3   // [b, h, dph, s]  (transposed V for attn@V)
#define HM_O    4   // [b, s, h*DPH+n] (batched z=b*H+h, m=s local)

// ---------------------------------------------------------------------------
// fp32 -> f16 conversion (weights, activation mirror)
// ---------------------------------------------------------------------------
__global__ void cvt_f2h(const float* __restrict__ src, _Float16* __restrict__ dst, int n) {
    int i = blockIdx.x * blockDim.x + threadIdx.x;
    if (i < n) dst[i] = (_Float16)src[i];
}

// ---------------------------------------------------------------------------
// Embedding: vocab_emb[raw] + sinusoidal PE + degree_emb[deg]  -> xf32
// ---------------------------------------------------------------------------
__global__ void embed_kernel(const int* __restrict__ raw, const int* __restrict__ deg,
                             const float* __restrict__ vemb, const float* __restrict__ demb,
                             float* __restrict__ x) {
    int i = blockIdx.x * blockDim.x + threadIdx.x;
    if (i >= M_ * D_) return;
    int d  = i % D_;
    int ms = i / D_;       // b*S + s
    int s  = ms % S_;
    int j  = d >> 1;
    float ang = (float)s * __expf(-(float)(2 * j) * 9.2103403719761836f / (float)D_);
    float pe  = (d & 1) ? __cosf(ang) : __sinf(ang);
    x[i] = vemb[(size_t)raw[ms] * D_ + d] + pe + demb[(size_t)deg[ms] * D_ + d];
}

// ---------------------------------------------------------------------------
// Add active_rel_emb at the "now" positions (atomic: rows may coincide)
// ---------------------------------------------------------------------------
__global__ void relnow_kernel(const int* __restrict__ pcn, const int* __restrict__ sbn,
                              const float* __restrict__ arel, float* __restrict__ x) {
    int i = blockIdx.x * blockDim.x + threadIdx.x;
    if (i >= 2 * B_ * D_) return;
    int g = i / (B_ * D_);
    int r = i % (B_ * D_);
    int b = r / D_, d = r % D_;
    const int* nw = g ? sbn : pcn;
    int bb = nw[b * 2 + 0];
    int ss = nw[b * 2 + 1];
    atomicAdd(&x[((size_t)bb * S_ + ss) * D_ + d], arel[g * D_ + d]);
}

// ---------------------------------------------------------------------------
// Scatter relation biases into the score tensor (per head, both directions)
// ---------------------------------------------------------------------------
__global__ void sbias_kernel(const int* __restrict__ pc, const int* __restrict__ sb,
                             const float* __restrict__ rel, float* __restrict__ scores) {
    int i = blockIdx.x * blockDim.x + threadIdx.x;
    if (i >= (NPC_ + NS_) * H_) return;
    int e = i / H_;
    int h = i % H_;
    int b, ii, jj;
    float f, g;
    if (e < NPC_) {
        b = pc[e * 3]; ii = pc[e * 3 + 1]; jj = pc[e * 3 + 2];
        f = rel[0]; g = rel[1];
    } else {
        e -= NPC_;
        b = sb[e * 3]; ii = sb[e * 3 + 1]; jj = sb[e * 3 + 2];
        f = rel[2]; g = rel[2];
    }
    size_t base = (size_t)(b * H_ + h) * S_ * S_;
    atomicAdd(&scores[base + (size_t)ii * S_ + jj], f);
    atomicAdd(&scores[base + (size_t)jj * S_ + ii], g);
}

// ---------------------------------------------------------------------------
// Softmax over last dim (512), fp32 in -> f16 out. One 128-thread block per row.
// ---------------------------------------------------------------------------
__global__ void softmax_kernel(const float* __restrict__ scores, _Float16* __restrict__ attn) {
    __shared__ float red[128];
    int row = blockIdx.x;
    int t   = threadIdx.x;
    const float* p = scores + (size_t)row * S_;
    float v[4];
    float mx = -3.4e38f;
#pragma unroll
    for (int u = 0; u < 4; u++) { v[u] = p[t + u * 128]; mx = fmaxf(mx, v[u]); }
    red[t] = mx; __syncthreads();
    for (int o = 64; o > 0; o >>= 1) { if (t < o) red[t] = fmaxf(red[t], red[t + o]); __syncthreads(); }
    mx = red[0]; __syncthreads();
    float sm = 0.f;
#pragma unroll
    for (int u = 0; u < 4; u++) { v[u] = __expf(v[u] - mx); sm += v[u]; }
    red[t] = sm; __syncthreads();
    for (int o = 64; o > 0; o >>= 1) { if (t < o) red[t] += red[t + o]; __syncthreads(); }
    float inv = 1.0f / red[0];
    _Float16* q = attn + (size_t)row * S_;
#pragma unroll
    for (int u = 0; u < 4; u++) q[t + u * 128] = (_Float16)(v[u] * inv);
}

// ---------------------------------------------------------------------------
// Fused residual + LayerNorm: x = LN(xin + c) -> xf32, xf16 (in-place safe)
// ---------------------------------------------------------------------------
__global__ void resln_kernel(const float* __restrict__ xin, const float* __restrict__ c,
                             const float* __restrict__ g, const float* __restrict__ bta,
                             float* __restrict__ xout, _Float16* __restrict__ xh) {
    __shared__ float red[128];
    int row = blockIdx.x;
    int t   = threadIdx.x;
    const float* xp = xin + (size_t)row * D_;
    const float* cp = c   + (size_t)row * D_;
    float v[4];
    float sm = 0.f;
#pragma unroll
    for (int u = 0; u < 4; u++) { v[u] = xp[t + u * 128] + cp[t + u * 128]; sm += v[u]; }
    red[t] = sm; __syncthreads();
    for (int o = 64; o > 0; o >>= 1) { if (t < o) red[t] += red[t + o]; __syncthreads(); }
    float mean = red[0] * (1.0f / D_); __syncthreads();
    float s2 = 0.f;
#pragma unroll
    for (int u = 0; u < 4; u++) { float d = v[u] - mean; s2 += d * d; }
    red[t] = s2; __syncthreads();
    for (int o = 64; o > 0; o >>= 1) { if (t < o) red[t] += red[t + o]; __syncthreads(); }
    float rs = rsqrtf(red[0] * (1.0f / D_) + 1e-5f);
#pragma unroll
    for (int u = 0; u < 4; u++) {
        int d = t + u * 128;
        float y = (v[u] - mean) * rs * g[d] + bta[d];
        xout[(size_t)row * D_ + d] = y;
        xh[(size_t)row * D_ + d]   = (_Float16)y;
    }
}

// ---------------------------------------------------------------------------
// Generic WMMA GEMM:  C[M,N] = A[M,K] @ W[N,K]^T  (f16 in, fp32 accumulate)
// One wave -> 32(M) x 64(N): 2 M-subtiles x 4 N-subtiles = 8 accumulators;
// each B fragment reused by 2 WMMAs -> 12 b128 loads per 8 WMMAs.
// Ping-pong double buffering with the K loop unrolled by two: fragments for
// even/odd K-steps live in statically distinct register sets, so there are
// no register copies and no WMMA->VALU WAR hazard NOPs in the steady state.
// Assumes K is a multiple of 64 or equal to 64 (true here: K in {64,512,2048}).
// Block = 8 waves arranged 4(M) x 2(N) = 128x128 tile.
// ---------------------------------------------------------------------------
__global__ __launch_bounds__(256) void gemm16(
    const _Float16* __restrict__ A, long long aStr, int lda,
    const _Float16* __restrict__ W, long long wStr, int ldw,
    const float* __restrict__ bias,
    float* __restrict__ Cout, long long cStr, int ldc, float scale,
    _Float16* __restrict__ Hout, int hmode, int ldh,
    int M, int N, int K, int relu)
{
    int z = blockIdx.z;
    A += (size_t)z * aStr;
    W += (size_t)z * wStr;

    int wave = threadIdx.x >> 5;
    int lane = threadIdx.x & 31;
    int lrow = lane & 15;
    int hi   = lane >> 4;                       // 0 or 1

    int mBase = blockIdx.y * 128 + (wave & 3) * 32;
    int nBase = blockIdx.x * 128 + (wave >> 2) * 64;

    const _Float16* aRow[2];
#pragma unroll
    for (int mt = 0; mt < 2; mt++) {
        int r = mBase + mt * 16 + lrow;
        if (r >= M) r = M - 1;                  // clamp; stores are predicated
        aRow[mt] = A + (size_t)r * lda + hi * 8;
    }
    const _Float16* wRow[4];
#pragma unroll
    for (int t = 0; t < 4; t++) {
        int n = nBase + t * 16 + lrow;
        if (n >= N) n = N - 1;
        wRow[t] = W + (size_t)n * ldw + hi * 8;
    }

    v8f acc[2][4];
#pragma unroll
    for (int mt = 0; mt < 2; mt++)
#pragma unroll
        for (int t = 0; t < 4; t++)
#pragma unroll
            for (int r = 0; r < 8; r++) acc[mt][t][r] = 0.f;

    // fragment load / compute helpers (pure register ops, inlined)
    auto loadA = [&](v16h (&a)[2], int kb) {
#pragma unroll
        for (int mt = 0; mt < 2; mt++) {
            v8h lo = *(const v8h*)(aRow[mt] + kb);
            v8h hh = *(const v8h*)(aRow[mt] + kb + 16);
#pragma unroll
            for (int i = 0; i < 8; i++) { a[mt][i] = lo[i]; a[mt][i + 8] = hh[i]; }
        }
    };
    auto loadB = [&](v16h (&b)[4], int kb) {
#pragma unroll
        for (int t = 0; t < 4; t++) {
            v8h lo = *(const v8h*)(wRow[t] + kb);
            v8h hh = *(const v8h*)(wRow[t] + kb + 16);
#pragma unroll
            for (int i = 0; i < 8; i++) { b[t][i] = lo[i]; b[t][i + 8] = hh[i]; }
        }
    };
    auto mm = [&](v16h (&a)[2], v16h (&b)[4]) {
#pragma unroll
        for (int mt = 0; mt < 2; mt++)
#pragma unroll
            for (int t = 0; t < 4; t++)
                acc[mt][t] = __builtin_amdgcn_wmma_f32_16x16x32_f16(
                    false, a[mt], false, b[t], (short)0, acc[mt][t], false, false);
    };

    // ---- ping-pong pipeline over K-steps of 32 (even step count) ----
    v16h a0[2], b0[4], a1[2], b1[4];
    int nk = K >> 5;                 // number of 32-wide K-steps (even)
    loadA(a0, 0);  loadB(b0, 0);
    if (nk > 1) { loadA(a1, 32); loadB(b1, 32); }
    int kb = 64;
    for (int s = 2; s < nk; s += 2) {
        __builtin_prefetch((const void*)(aRow[0] + kb + 64), 0, 1);  // speculative
        mm(a0, b0);
        loadA(a0, kb);      loadB(b0, kb);
        mm(a1, b1);
        loadA(a1, kb + 32); loadB(b1, kb + 32);
        kb += 64;
    }
    mm(a0, b0);
    if (nk > 1) mm(a1, b1);

    // ---- store. C/D layout: VGPR r -> row sub + hi*8 + r, col = lrow ----
#pragma unroll
    for (int mt = 0; mt < 2; mt++) {
#pragma unroll
        for (int t = 0; t < 4; t++) {
            int n = nBase + t * 16 + lrow;
            if (n >= N) continue;
            float bvv = bias ? bias[n] : 0.f;
#pragma unroll
            for (int r = 0; r < 8; r++) {
                int m = mBase + mt * 16 + hi * 8 + r;
                if (m >= M) continue;
                float val = acc[mt][t][r] * scale + bvv;
                if (relu) val = fmaxf(val, 0.f);
                if (Cout) Cout[(size_t)z * cStr + (size_t)m * ldc + n] = val;
                if (Hout) {
                    size_t hidx;
                    if (hmode == HM_ROW) {
                        hidx = (size_t)m * ldh + n;
                    } else if (hmode == HM_QK) {
                        int b = m / S_, s = m % S_, h = n / DPH_, dp = n % DPH_;
                        hidx = ((size_t)(b * H_ + h) * S_ + s) * DPH_ + dp;
                    } else if (hmode == HM_VT) {
                        int b = m / S_, s = m % S_, h = n / DPH_, dp = n % DPH_;
                        hidx = ((size_t)(b * H_ + h) * DPH_ + dp) * S_ + s;
                    } else { // HM_O
                        int b = z / H_, h = z % H_;
                        hidx = (size_t)(b * S_ + m) * D_ + h * DPH_ + n;
                    }
                    Hout[hidx] = (_Float16)val;
                }
            }
        }
    }
}

// ---------------------------------------------------------------------------
// Host-side orchestration
// ---------------------------------------------------------------------------
extern "C" void kernel_launch(void* const* d_in, const int* in_sizes, int n_in,
                              void* d_out, int out_size, void* d_ws, size_t ws_size,
                              hipStream_t stream) {
    (void)in_sizes; (void)n_in; (void)out_size; (void)ws_size;

    const int*   raw  = (const int*)d_in[0];
    const int*   deg  = (const int*)d_in[1];
    const int*   pc   = (const int*)d_in[2];
    const int*   pcn  = (const int*)d_in[3];
    const int*   sb   = (const int*)d_in[4];
    const int*   sbn  = (const int*)d_in[5];
    const float* vemb = (const float*)d_in[6];
    const float* demb = (const float*)d_in[7];
    const float* arel = (const float*)d_in[8];
    const float* rel  = (const float*)d_in[9];
    const float* Wq   = (const float*)d_in[10]; const float* bq = (const float*)d_in[11];
    const float* Wk   = (const float*)d_in[12]; const float* bk = (const float*)d_in[13];
    const float* Wv   = (const float*)d_in[14]; const float* bv = (const float*)d_in[15];
    const float* Wo   = (const float*)d_in[16]; const float* bo = (const float*)d_in[17];
    const float* W1   = (const float*)d_in[18]; const float* b1 = (const float*)d_in[19];
    const float* W2   = (const float*)d_in[20]; const float* b2 = (const float*)d_in[21];
    const float* lng  = (const float*)d_in[22]; const float* lnb = (const float*)d_in[23];
    const float* lw   = (const float*)d_in[24];

    // ----- workspace layout (256B aligned slices) -----
    char* ws = (char*)d_ws;
    size_t off = 0;
    auto alloc = [&](size_t bytes) -> char* {
        char* p = ws + off;
        off += (bytes + 255) & ~(size_t)255;
        return p;
    };
    _Float16* wq16 = (_Float16*)alloc((size_t)L_ * D_ * D_ * 2);
    _Float16* wk16 = (_Float16*)alloc((size_t)L_ * D_ * D_ * 2);
    _Float16* wv16 = (_Float16*)alloc((size_t)L_ * D_ * D_ * 2);
    _Float16* wo16 = (_Float16*)alloc((size_t)L_ * D_ * D_ * 2);
    _Float16* w116 = (_Float16*)alloc((size_t)L_ * DFF_ * D_ * 2);
    _Float16* w216 = (_Float16*)alloc((size_t)L_ * D_ * DFF_ * 2);
    _Float16* lg16 = (_Float16*)alloc((size_t)VOCAB_ * D_ * 2);
    float*    xf32 = (float*)   alloc((size_t)M_ * D_ * 4);
    _Float16* xf16 = (_Float16*)alloc((size_t)M_ * D_ * 2);
    _Float16* q16  = (_Float16*)alloc((size_t)M_ * D_ * 2);
    _Float16* k16  = (_Float16*)alloc((size_t)M_ * D_ * 2);
    _Float16* vt16 = (_Float16*)alloc((size_t)M_ * D_ * 2);
    float*    scor = (float*)   alloc((size_t)B_ * H_ * S_ * S_ * 4);
    _Float16* at16 = (_Float16*)alloc((size_t)B_ * H_ * S_ * S_ * 2);
    _Float16* o16  = (_Float16*)alloc((size_t)M_ * D_ * 2);
    _Float16* h116 = (_Float16*)alloc((size_t)M_ * DFF_ * 2);
    float*    Cbuf = (float*)   alloc((size_t)M_ * D_ * 4);

    auto cvt = [&](const float* s, _Float16* d, size_t n) {
        cvt_f2h<<<(int)((n + 255) / 256), 256, 0, stream>>>(s, d, (int)n);
    };
    auto gemm = [&](const _Float16* A, long long aS, int lda,
                    const _Float16* Wt, long long wS, int ldw,
                    const float* bias_, float* C, long long cS, int ldc, float scale,
                    _Float16* Hh, int hmode, int ldh,
                    int Mm, int Nn, int Kk, int relu, int Z) {
        dim3 g((Nn + 127) / 128, (Mm + 127) / 128, Z);
        gemm16<<<g, 256, 0, stream>>>(A, aS, lda, Wt, wS, ldw, bias_,
                                      C, cS, ldc, scale, Hh, hmode, ldh,
                                      Mm, Nn, Kk, relu);
    };

    // ----- weights -> f16 (once per launch, deterministic) -----
    cvt(Wq, wq16, (size_t)L_ * D_ * D_);
    cvt(Wk, wk16, (size_t)L_ * D_ * D_);
    cvt(Wv, wv16, (size_t)L_ * D_ * D_);
    cvt(Wo, wo16, (size_t)L_ * D_ * D_);
    cvt(W1, w116, (size_t)L_ * DFF_ * D_);
    cvt(W2, w216, (size_t)L_ * D_ * DFF_);
    cvt(lw, lg16, (size_t)VOCAB_ * D_);

    // ----- embeddings + relation-now adds, then f16 mirror -----
    embed_kernel<<<(M_ * D_ + 255) / 256, 256, 0, stream>>>(raw, deg, vemb, demb, xf32);
    relnow_kernel<<<(2 * B_ * D_ + 255) / 256, 256, 0, stream>>>(pcn, sbn, arel, xf32);
    cvt(xf32, xf16, (size_t)M_ * D_);

    // ----- transformer layers -----
    for (int l = 0; l < L_; l++) {
        size_t wOff  = (size_t)l * D_ * D_;
        size_t w1Off = (size_t)l * DFF_ * D_;
        size_t w2Off = (size_t)l * D_ * DFF_;

        // Q, K (head layout), V (transposed head layout)
        gemm(xf16, 0, D_, wq16 + wOff, 0, D_, bq + (size_t)l * D_,
             nullptr, 0, 0, 1.f, q16, HM_QK, 0, M_, D_, D_, 0, 1);
        gemm(xf16, 0, D_, wk16 + wOff, 0, D_, bk + (size_t)l * D_,
             nullptr, 0, 0, 1.f, k16, HM_QK, 0, M_, D_, D_, 0, 1);
        gemm(xf16, 0, D_, wv16 + wOff, 0, D_, bv + (size_t)l * D_,
             nullptr, 0, 0, 1.f, vt16, HM_VT, 0, M_, D_, D_, 0, 1);

        // scores = q @ k^T * 1/sqrt(DPH), batched over z = b*H+h
        gemm(q16, (long long)S_ * DPH_, DPH_, k16, (long long)S_ * DPH_, DPH_,
             nullptr, scor, (long long)S_ * S_, S_, 0.125f,
             nullptr, HM_NONE, 0, S_, S_, DPH_, 0, B_ * H_);

        // relation bias scatters, softmax -> f16 attention
        sbias_kernel<<<((NPC_ + NS_) * H_ + 255) / 256, 256, 0, stream>>>(pc, sb, rel, scor);
        softmax_kernel<<<B_ * H_ * S_, 128, 0, stream>>>(scor, at16);

        // o = attn @ v  (writes straight into [b,s,d] f16)
        gemm(at16, (long long)S_ * S_, S_, vt16, (long long)DPH_ * S_, S_,
             nullptr, nullptr, 0, 0, 1.f, o16, HM_O, 0, S_, DPH_, S_, 0, B_ * H_);

        // output projection -> Cbuf, then x = LN(x + Cbuf)
        gemm(o16, 0, D_, wo16 + wOff, 0, D_, bo + (size_t)l * D_,
             Cbuf, 0, D_, 1.f, nullptr, HM_NONE, 0, M_, D_, D_, 0, 1);
        resln_kernel<<<M_, 128, 0, stream>>>(xf32, Cbuf, lng + (size_t)l * D_,
                                             lnb + (size_t)l * D_, xf32, xf16);

        // FFN: h1 = relu(x@W1^T+b1) f16 ; C = h1@W2^T+b2 ; x = LN(x + C)
        gemm(xf16, 0, D_, w116 + w1Off, 0, D_, b1 + (size_t)l * DFF_,
             nullptr, 0, 0, 1.f, h116, HM_ROW, DFF_, M_, DFF_, D_, 1, 1);
        gemm(h116, 0, DFF_, w216 + w2Off, 0, DFF_, b2 + (size_t)l * D_,
             Cbuf, 0, D_, 1.f, nullptr, HM_NONE, 0, M_, D_, DFF_, 0, 1);
        resln_kernel<<<M_, 128, 0, stream>>>(xf32, Cbuf, lng + (size_t)l * D_,
                                             lnb + (size_t)l * D_, xf32, xf16);
    }

    // ----- logits: x[:, -1] @ logits_w^T  (lda trick selects s = S-1 rows) -----
    gemm(xf16 + (size_t)(S_ - 1) * D_, 0, S_ * D_, lg16, 0, D_,
         nullptr, (float*)d_out, 0, VOCAB_, 1.f,
         nullptr, HM_NONE, 0, B_, VOCAB_, D_, 0, 1);
}
